// EncoderLayer_11879879543805
// MI455X (gfx1250) — matrix operand order repaired and checked by reading, not compiled, not run on hardware
//
#include <hip/hip_runtime.h>
#include <hip/hip_bf16.h>

// CDNA5 / gfx1250 fused encoder-layer kernel:
//   qkv = x @ W            (bf16 WMMA, f32 accum)   [1024, 64, 512]
//   s   = (qkv qkv^T) * mask                         [1024, 64, 64]
//   out = s @ qkv                                    [1024, 64, 512]
// One workgroup (8 wave32) per batch; qkv and scores live only in LDS.

typedef __bf16 bf16_t;
typedef __attribute__((ext_vector_type(16))) __bf16 v16bf;
typedef __attribute__((ext_vector_type(8)))  __bf16 v8bf;
typedef __attribute__((ext_vector_type(8)))  float  v8f;

#define D 512
#define S 64
#define QKV_STRIDE 520   // bf16 elems; 1040B rows -> 16-lane column reads hit distinct banks
#define QT_STRIDE  72    // bf16 elems; 144B rows  -> distinct banks
#define SC_STRIDE  72

// ---------------- helpers: ISA-layout fragment loads ----------------

// A-operand 16x32 bf16 (row = lane%16; K chunks split by lane-half), from bf16 row ptr.
__device__ inline v16bf load_a_bf16(const bf16_t* __restrict__ row, int k0, int lhalf) {
  const int kb = k0 + lhalf * 8;
  v8bf lo = *(const v8bf*)(row + kb);        // K = kb .. kb+7
  v8bf hi = *(const v8bf*)(row + kb + 16);   // K = kb+16 .. kb+23
  return __builtin_shufflevector(lo, hi, 0,1,2,3,4,5,6,7,8,9,10,11,12,13,14,15);
}

// A-operand from f32 row (global x), converting to bf16 on the fly.
__device__ inline v16bf load_a_f32(const float* __restrict__ row, int k0, int lhalf) {
  const int kb = k0 + lhalf * 8;
  const float4* p = (const float4*)(row + kb);        // 32B aligned
  const float4* q = (const float4*)(row + kb + 16);
  float4 c0 = p[0], c1 = p[1], c2 = q[0], c3 = q[1];
  v16bf a;
  a[0]=(bf16_t)c0.x; a[1]=(bf16_t)c0.y; a[2]=(bf16_t)c0.z; a[3]=(bf16_t)c0.w;
  a[4]=(bf16_t)c1.x; a[5]=(bf16_t)c1.y; a[6]=(bf16_t)c1.z; a[7]=(bf16_t)c1.w;
  a[8]=(bf16_t)c2.x; a[9]=(bf16_t)c2.y; a[10]=(bf16_t)c2.z; a[11]=(bf16_t)c2.w;
  a[12]=(bf16_t)c3.x; a[13]=(bf16_t)c3.y; a[14]=(bf16_t)c3.z; a[15]=(bf16_t)c3.w;
  return a;
}

// B-operand 32x16 bf16: lane = column, K = lhalf*8 + e (e<8), 16 + lhalf*8 + (e-8) (e>=8).
__device__ inline v16bf load_b(const bf16_t* __restrict__ colrow, int k0, int lhalf) {
  const bf16_t* p = colrow + k0 + lhalf * 8;
  v8bf lo = *(const v8bf*)(p);
  v8bf hi = *(const v8bf*)(p + 16);
  return __builtin_shufflevector(lo, hi, 0,1,2,3,4,5,6,7,8,9,10,11,12,13,14,15);
}

// ---------------- prep: W[k][n] f32 -> Wt[n][k] bf16 (coalesced tiled transpose) ----

__global__ void prep_w_kernel(const float* __restrict__ W, bf16_t* __restrict__ Wt) {
  __shared__ float tile[64][65];
  const int bk = (blockIdx.x & 7) * 64;   // k tile
  const int bn = (blockIdx.x >> 3) * 64;  // n tile
  for (int i = threadIdx.x; i < 64 * 64; i += 256) {
    int r = i >> 6, c = i & 63;           // r: k offset, c: n offset (coalesced read)
    tile[r][c] = W[(size_t)(bk + r) * D + bn + c];
  }
  __syncthreads();
  for (int i = threadIdx.x; i < 64 * 64; i += 256) {
    int n = i >> 6, k = i & 63;           // coalesced write along k
    Wt[(size_t)(bn + n) * D + bk + k] = (bf16_t)tile[k][n];
  }
}

// ---------------- fused per-batch kernel ----------------

__global__ void __launch_bounds__(256)
encoder_fused_kernel(const float* __restrict__ x,
                     const float* __restrict__ mask,
                     const bf16_t* __restrict__ Wt,
                     float* __restrict__ out) {
  extern __shared__ __align__(16) char smem[];
  bf16_t* qkv  = (bf16_t*)smem;                 // [64][QKV_STRIDE]
  bf16_t* qkvT = qkv  + S * QKV_STRIDE;         // [512][QT_STRIDE]
  bf16_t* sc   = qkvT + D * QT_STRIDE;          // [64][SC_STRIDE]

  const int b     = blockIdx.x;
  const int wave  = threadIdx.x >> 5;
  const int lane  = threadIdx.x & 31;
  const int lhalf = lane >> 4;                  // 0/1
  const int lmod  = lane & 15;

  // ---- phase 1: qkv = x_b @ W  (M=64, N=512, K=512) ----
  // wave -> mtile = wave&3 (rows), nhalf = wave>>2 (16 n-tiles each)
  const int mtile = wave & 3;
  const int nhalf = wave >> 2;

  v8f acc[16] = {};
  const float* xrow = x + ((size_t)b * S + mtile * 16 + lmod) * D;

  for (int ks = 0; ks < 16; ks++) {
    const int k0 = ks * 32;
    v16bf a = load_a_f32(xrow, k0, lhalf);
#pragma unroll
    for (int nt = 0; nt < 16; nt++) {
      const bf16_t* col = Wt + (size_t)((nhalf * 16 + nt) * 16 + lmod) * D;
      v16bf bb = load_b(col, k0, lhalf);
      acc[nt] = __builtin_amdgcn_wmma_f32_16x16x32_bf16(
          false, a, false, bb, (short)0, acc[nt], false, false);
    }
  }
  // write qkv (row-major, scalar b16) and qkvT (transposed, packed b128) to LDS.
  // C layout: VGPR r, lane -> M = mtile*16 + r + lhalf*8, N = tile_n*16 + lmod.
  // The 8 r-values per lane are consecutive M -> contiguous in qkvT[N][M]: one b128 store.
#pragma unroll
  for (int nt = 0; nt < 16; nt++) {
    const int N = (nhalf * 16 + nt) * 16 + lmod;
    v8bf pk;
#pragma unroll
    for (int r = 0; r < 8; r++) {
      bf16_t v = (bf16_t)acc[nt][r];
      qkv[(mtile * 16 + r + lhalf * 8) * QKV_STRIDE + N] = v;
      pk[r] = v;
    }
    *(v8bf*)(qkvT + N * QT_STRIDE + mtile * 16 + lhalf * 8) = pk;  // 16B aligned
  }
  __syncthreads();

  // ---- phase 2: scores = qkv qkv^T, masked  (M=64, N=64, K=512) ----
  // 16 tiles total; wave -> jtile = wave>>1, two ftiles = (wave&1)*2 + {0,1}
  {
    const int jtile = wave >> 1;
    const int ft0   = (wave & 1) * 2;
    v8f sacc[2] = {};
    const bf16_t* arow = qkv + (jtile * 16 + lmod) * QKV_STRIDE;

    for (int ks = 0; ks < 16; ks++) {
      const int k0 = ks * 32;
      v16bf a = load_a_bf16(arow, k0, lhalf);
#pragma unroll
      for (int tt = 0; tt < 2; tt++) {
        const bf16_t* brow = qkv + ((ft0 + tt) * 16 + lmod) * QKV_STRIDE;
        v16bf bb = load_b(brow, k0, lhalf);
        sacc[tt] = __builtin_amdgcn_wmma_f32_16x16x32_bf16(
            false, a, false, bb, (short)0, sacc[tt], false, false);
      }
    }
#pragma unroll
    for (int tt = 0; tt < 2; tt++) {
      const int f = (ft0 + tt) * 16 + lmod;
      const float m = mask[(size_t)b * S + f];   // mask over key axis
#pragma unroll
      for (int r = 0; r < 8; r++) {
        const int j = jtile * 16 + r + lhalf * 8;
        sc[j * SC_STRIDE + f] = (bf16_t)(sacc[tt][r] * m);
      }
    }
  }
  __syncthreads();

  // ---- phase 3: out = scores @ qkv  (M=64, N=512, K=64), B from qkvT ----
  {
    v8f oacc[16] = {};
    const bf16_t* arow = sc + (mtile * 16 + lmod) * SC_STRIDE;
    for (int ks = 0; ks < 2; ks++) {
      const int k0 = ks * 32;
      v16bf a = load_a_bf16(arow, k0, lhalf);
#pragma unroll
      for (int nt = 0; nt < 16; nt++) {
        const bf16_t* brow = qkvT + (size_t)((nhalf * 16 + nt) * 16 + lmod) * QT_STRIDE;
        v16bf bb = load_b(brow, k0, lhalf);
        oacc[nt] = __builtin_amdgcn_wmma_f32_16x16x32_bf16(
            false, a, false, bb, (short)0, oacc[nt], false, false);
      }
    }
    float* ob = out + (size_t)b * S * D;
#pragma unroll
    for (int nt = 0; nt < 16; nt++) {
      const int N = (nhalf * 16 + nt) * 16 + lmod;
#pragma unroll
      for (int r = 0; r < 8; r++) {
        const int M = mtile * 16 + r + lhalf * 8;
        ob[M * D + N] = oacc[nt][r];
      }
    }
  }
}

// ---------------- host launch ----------------

extern "C" void kernel_launch(void* const* d_in, const int* in_sizes, int n_in,
                              void* d_out, int out_size, void* d_ws, size_t ws_size,
                              hipStream_t stream) {
  const float* x    = (const float*)d_in[0];   // [1024, 64, 512] f32
  const float* mask = (const float*)d_in[1];   // [1024, 64]      f32
  const float* W    = (const float*)d_in[2];   // [512, 512]      f32
  float* out = (float*)d_out;                  // [1024, 64, 512] f32
  bf16_t* Wt = (bf16_t*)d_ws;                  // 512 KB transposed bf16 W

  const size_t smem = (size_t)(S * QKV_STRIDE + D * QT_STRIDE + S * SC_STRIDE) * sizeof(bf16_t);
  // ~146 KB dynamic LDS: lift the 64 KB default cap (WGP supports 320 KB).
  hipFuncSetAttribute((const void*)encoder_fused_kernel,
                      hipFuncAttributeMaxDynamicSharedMemorySize, (int)smem);

  prep_w_kernel<<<64, 256, 0, stream>>>(W, Wt);
  encoder_fused_kernel<<<1024, 256, smem, stream>>>(x, mask, Wt, out);
}
